// LinearAttentionMechanism_16612933501288
// MI455X (gfx1250) — matrix operand order repaired and checked by reading, not compiled, not run on hardware
//
#include <hip/hip_runtime.h>
#include <hip/hip_bf16.h>

// ---- CDNA5 (gfx1250) wave32 WMMA implementation of the linear-attention block ----
// All heavy math: v_wmma_f32_16x16x32_bf16. All GEMM operands pre-converted to
// bf16 so tile staging can use GLOBAL_LOAD_ASYNC_TO_LDS_B128 (ASYNCcnt path)
// with 2-deep LDS double buffering.

typedef __bf16 bf16_t;
typedef __attribute__((ext_vector_type(16))) __bf16 v16bf;
typedef __attribute__((ext_vector_type(8)))  __bf16 v8bf;
typedef __attribute__((ext_vector_type(8)))  float  v8f;
typedef int v4i __attribute__((vector_size(16)));

#define DIM_ 1024
#define NH_  16
#define HD_  64
#define B_   4
#define L_   4096

#if __has_builtin(__builtin_amdgcn_global_load_async_to_lds_b128)
#define ASYNC_LDS 1
typedef __attribute__((address_space(1))) v4i* gv4i_p;
typedef __attribute__((address_space(3))) v4i* lv4i_p;
#endif

static __device__ inline void cp16_async(const bf16_t* __restrict__ g, bf16_t* __restrict__ l) {
#ifdef ASYNC_LDS
  __builtin_amdgcn_global_load_async_to_lds_b128(
      (gv4i_p)(__attribute__((address_space(1))) void*)g,
      (lv4i_p)(__attribute__((address_space(3))) void*)l, 0, 0);
#else
  *reinterpret_cast<v8bf*>(l) = *reinterpret_cast<const v8bf*>(g);
#endif
}

static __device__ inline void wait_async() {
#ifdef ASYNC_LDS
#if __has_builtin(__builtin_amdgcn_s_wait_asynccnt)
  __builtin_amdgcn_s_wait_asynccnt(0);
#else
  asm volatile("s_wait_asynccnt 0x0" ::: "memory");
#endif
#endif
}

static __device__ inline v16bf mk16(v8bf lo, v8bf hi) {
  v16bf r;
#pragma unroll
  for (int i = 0; i < 8; ++i) { r[i] = lo[i]; r[i + 8] = hi[i]; }
  return r;
}

static __device__ inline v8f wmma_bf16(v16bf a, v16bf b, v8f c) {
  return __builtin_amdgcn_wmma_f32_16x16x32_bf16(
      false, a, false, b, (short)0, c, false, false);
}

// ---------------------------------------------------------------------------
// fp32 -> bf16 conversion pass (8 elements / thread, vectorized)
// ---------------------------------------------------------------------------
__launch_bounds__(256)
__global__ void to_bf16(const float* __restrict__ src, bf16_t* __restrict__ dst) {
  const size_t i = ((size_t)blockIdx.x * 256 + threadIdx.x) * 8;
  float4 a = *reinterpret_cast<const float4*>(src + i);
  float4 b = *reinterpret_cast<const float4*>(src + i + 4);
  v8bf o;
  o[0] = (bf16_t)a.x; o[1] = (bf16_t)a.y; o[2] = (bf16_t)a.z; o[3] = (bf16_t)a.w;
  o[4] = (bf16_t)b.x; o[5] = (bf16_t)b.y; o[6] = (bf16_t)b.z; o[7] = (bf16_t)b.w;
  *reinterpret_cast<v8bf*>(dst + i) = o;
}

// ---------------------------------------------------------------------------
// Y[M,N] = A[M,Kd] * W[N,Kd]^T + bias.  A, W bf16; Y fp32 or bf16.
// BM=128, BN=128, BK=64; 256 threads = 8 waves (4m x 2n), each wave 32x64.
// Double-buffered LDS, async-to-LDS staging.
// ---------------------------------------------------------------------------
template <bool OUT_BF16>
__launch_bounds__(256)
__global__ void gemm_bf16k(const bf16_t* __restrict__ A, const bf16_t* __restrict__ W,
                           const float* __restrict__ bias, void* __restrict__ Yv,
                           int M, int N, int Kd) {
  __shared__ alignas(16) bf16_t As[2][128 * 64];
  __shared__ alignas(16) bf16_t Bs[2][128 * 64];

  const int tid  = threadIdx.x;
  const int lane = tid & 31;
  const int wid  = tid >> 5;
  const int wm   = wid & 3;   // m strip: rows 32*wm
  const int wn   = wid >> 2;  // n strip: cols 64*wn
  const int m0   = blockIdx.y * 128;
  const int n0   = blockIdx.x * 128;

  const int srow = tid >> 1;        // 0..127
  const int scol = (tid & 1) * 32;  // 0 or 32

  const bf16_t* gA = A + (size_t)(m0 + srow) * Kd + scol;
  const bf16_t* gW = W + (size_t)(n0 + srow) * Kd + scol;
  bf16_t* lA = &As[0][srow * 64 + scol];
  bf16_t* lB = &Bs[0][srow * 64 + scol];
  const int lstride = 128 * 64;  // buffer stride in elements

  v8f acc[2][4];
#pragma unroll
  for (int i = 0; i < 2; ++i)
#pragma unroll
    for (int j = 0; j < 4; ++j) acc[i][j] = {};

  // prologue: stage tile 0 into buffer 0
#pragma unroll
  for (int c = 0; c < 4; ++c) {
    cp16_async(gA + c * 8, lA + c * 8);
    cp16_async(gW + c * 8, lB + c * 8);
  }

  const int ntiles = Kd >> 6;
  for (int t = 0; t < ntiles; ++t) {
    wait_async();
    __syncthreads();  // tile t visible to all waves; all waves done reading tile t-1's buffer

    const int cur = t & 1;
    if (t + 1 < ntiles) {
      const int nxt = cur ^ 1;
      const int ko  = (t + 1) * 64;
#pragma unroll
      for (int c = 0; c < 4; ++c) {
        cp16_async(gA + ko + c * 8, lA + nxt * lstride + c * 8);
        cp16_async(gW + ko + c * 8, lB + nxt * lstride + c * 8);
      }
    }

#pragma unroll
    for (int kk = 0; kk < 64; kk += 32) {
      v16bf a[2], b[4];
#pragma unroll
      for (int i = 0; i < 2; ++i) {
        // A frag 16x32: lane -> row lane%16, K chunks 8*(lane/16) and +16
        const bf16_t* p = &As[cur][(wm * 32 + i * 16 + (lane & 15)) * 64 + kk + 8 * (lane >> 4)];
        a[i] = mk16(*(const v8bf*)p, *(const v8bf*)(p + 16));
      }
#pragma unroll
      for (int j = 0; j < 4; ++j) {
        // B frag 32x16: lane -> col lane%16, K chunks 8*(lane/16) and +16
        const bf16_t* p = &Bs[cur][(wn * 64 + j * 16 + (lane & 15)) * 64 + kk + 8 * (lane >> 4)];
        b[j] = mk16(*(const v8bf*)p, *(const v8bf*)(p + 16));
      }
#pragma unroll
      for (int i = 0; i < 2; ++i)
#pragma unroll
        for (int j = 0; j < 4; ++j)
          acc[i][j] = wmma_bf16(a[i], b[j], acc[i][j]);
    }
  }

  // epilogue: C/D layout -> row = v + 8*(lane/16), col = lane%16
#pragma unroll
  for (int j = 0; j < 4; ++j) {
    const int n    = n0 + wn * 64 + j * 16 + (lane & 15);
    const float bb = bias[n];
#pragma unroll
    for (int i = 0; i < 2; ++i) {
      const int mb = m0 + wm * 32 + i * 16 + 8 * (lane >> 4);
#pragma unroll
      for (int v = 0; v < 8; ++v) {
        const float val  = acc[i][j][v] + bb;
        const size_t idx = (size_t)(mb + v) * N + n;
        if constexpr (OUT_BF16) ((bf16_t*)Yv)[idx] = (bf16_t)val;
        else                    ((float*)Yv)[idx]  = val;
      }
    }
  }
}

// ---------------------------------------------------------------------------
// In-place softmax over the L dimension of K[B, L, DIM] (one block per column)
// ---------------------------------------------------------------------------
__launch_bounds__(256)
__global__ void softmax_col(bf16_t* __restrict__ Kb) {
  const int c = blockIdx.x;
  const int b = blockIdx.y;
  const int tid = threadIdx.x;
  __shared__ float red[256];

  float v[16];
  float mx = -1e30f;
#pragma unroll
  for (int i = 0; i < 16; ++i) {
    const int l = tid + i * 256;
    v[i] = (float)Kb[((size_t)(b * L_ + l)) * DIM_ + c];
    mx = fmaxf(mx, v[i]);
  }
  red[tid] = mx;
  __syncthreads();
  for (int s = 128; s > 0; s >>= 1) {
    if (tid < s) red[tid] = fmaxf(red[tid], red[tid + s]);
    __syncthreads();
  }
  mx = red[0];
  __syncthreads();

  float sum = 0.f;
#pragma unroll
  for (int i = 0; i < 16; ++i) { v[i] = __expf(v[i] - mx); sum += v[i]; }
  red[tid] = sum;
  __syncthreads();
  for (int s = 128; s > 0; s >>= 1) {
    if (tid < s) red[tid] += red[tid + s];
    __syncthreads();
  }
  const float inv = 1.f / red[0];
#pragma unroll
  for (int i = 0; i < 16; ++i) {
    const int l = tid + i * 256;
    Kb[((size_t)(b * L_ + l)) * DIM_ + c] = (bf16_t)(v[i] * inv);
  }
}

// ---------------------------------------------------------------------------
// ctxT[b,h][e][d] = sum_l Kp[b,l,h*64+d] * V[b,l,h*64+e]
// One block per (b,h); both operands staged transposed into LDS so A/B
// fragments read contiguous K(=l).
// ---------------------------------------------------------------------------
__launch_bounds__(128)
__global__ void context_kernel(const bf16_t* __restrict__ Kp,
                               const bf16_t* __restrict__ V,
                               bf16_t* __restrict__ CtxT) {
  const int h = blockIdx.x;
  const int b = blockIdx.y;
  const int tid  = threadIdx.x;
  const int lane = tid & 31;
  const int w    = tid >> 5;

  __shared__ alignas(16) bf16_t KT[64 * 64];  // [d][l]
  __shared__ alignas(16) bf16_t VT[64 * 64];  // [e][l]

  v8f acc[4];
#pragma unroll
  for (int j = 0; j < 4; ++j) acc[j] = {};

  const int lloc = tid >> 1;
  const int c0   = (tid & 1) * 32;

  for (int l0 = 0; l0 < L_; l0 += 64) {
    const size_t gbase = ((size_t)(b * L_ + l0 + lloc)) * DIM_ + h * HD_ + c0;
    const v8bf* gk = reinterpret_cast<const v8bf*>(Kp + gbase);
    const v8bf* gv = reinterpret_cast<const v8bf*>(V + gbase);
#pragma unroll
    for (int jj = 0; jj < 4; ++jj) {
      v8bf xk = gk[jj], xv = gv[jj];
#pragma unroll
      for (int e = 0; e < 8; ++e) {
        KT[(c0 + jj * 8 + e) * 64 + lloc] = xk[e];
        VT[(c0 + jj * 8 + e) * 64 + lloc] = xv[e];
      }
    }
    __syncthreads();

#pragma unroll
    for (int kk = 0; kk < 64; kk += 32) {
      const bf16_t* pa = KT + (w * 16 + (lane & 15)) * 64 + kk + 8 * (lane >> 4);
      v16bf a = mk16(*(const v8bf*)pa, *(const v8bf*)(pa + 16));
#pragma unroll
      for (int j = 0; j < 4; ++j) {
        const bf16_t* pb = VT + (j * 16 + (lane & 15)) * 64 + kk + 8 * (lane >> 4);
        v16bf bb = mk16(*(const v8bf*)pb, *(const v8bf*)(pb + 16));
        acc[j] = wmma_bf16(a, bb, acc[j]);
      }
    }
    __syncthreads();
  }

  bf16_t* out = CtxT + ((size_t)(b * NH_ + h)) * 64 * 64;
#pragma unroll
  for (int j = 0; j < 4; ++j) {
    const int e = j * 16 + (lane & 15);
#pragma unroll
    for (int v = 0; v < 8; ++v) {
      const int d = w * 16 + v + 8 * (lane >> 4);
      out[e * 64 + d] = (bf16_t)acc[j][v];
    }
  }
}

// ---------------------------------------------------------------------------
// Ab[b, l, h*64+e] = sum_d Q[b,l,h*64+d] * ctx[b,h][d][e]   (bf16 out)
// ---------------------------------------------------------------------------
__launch_bounds__(128)
__global__ void attended_kernel(const bf16_t* __restrict__ Q,
                                const bf16_t* __restrict__ CtxT,
                                bf16_t* __restrict__ Ab) {
  const int l0 = blockIdx.x * 64;
  const int h  = blockIdx.y & (NH_ - 1);
  const int b  = blockIdx.y >> 4;
  const int tid  = threadIdx.x;
  const int lane = tid & 31;
  const int w    = tid >> 5;

  v8f acc[4];
#pragma unroll
  for (int j = 0; j < 4; ++j) acc[j] = {};

  const bf16_t* ctx = CtxT + ((size_t)(b * NH_ + h)) * 64 * 64;

#pragma unroll
  for (int kk = 0; kk < 64; kk += 32) {
    const int row = l0 + w * 16 + (lane & 15);
    const bf16_t* pa = Q + ((size_t)(b * L_ + row)) * DIM_ + h * HD_ + kk + 8 * (lane >> 4);
    v16bf a = mk16(*(const v8bf*)pa, *(const v8bf*)(pa + 16));
#pragma unroll
    for (int j = 0; j < 4; ++j) {
      const bf16_t* pb = ctx + (j * 16 + (lane & 15)) * 64 + kk + 8 * (lane >> 4);
      v16bf bb = mk16(*(const v8bf*)pb, *(const v8bf*)(pb + 16));
      acc[j] = wmma_bf16(a, bb, acc[j]);
    }
  }

#pragma unroll
  for (int j = 0; j < 4; ++j) {
    const int e = j * 16 + (lane & 15);
#pragma unroll
    for (int v = 0; v < 8; ++v) {
      const int row = l0 + w * 16 + v + 8 * (lane >> 4);
      Ab[((size_t)(b * L_ + row)) * DIM_ + h * HD_ + e] = (bf16_t)acc[j][v];
    }
  }
}

// ---------------------------------------------------------------------------
extern "C" void kernel_launch(void* const* d_in, const int* in_sizes, int n_in,
                              void* d_out, int out_size, void* d_ws, size_t ws_size,
                              hipStream_t stream) {
  (void)in_sizes; (void)n_in; (void)out_size; (void)ws_size;

  const float* q_seq = (const float*)d_in[0];
  const float* k_seq = (const float*)d_in[1];
  const float* v_seq = (const float*)d_in[2];
  const float* Wq = (const float*)d_in[3];
  const float* bq = (const float*)d_in[4];
  const float* Wk = (const float*)d_in[5];
  const float* bk = (const float*)d_in[6];
  const float* Wv = (const float*)d_in[7];
  const float* bv = (const float*)d_in[8];
  const float* Wo = (const float*)d_in[9];
  const float* bo = (const float*)d_in[10];
  // extract_common == 1 in setup_inputs(); "target = attended" path implemented.

  const int M = B_ * L_;              // 16384
  const size_t S = (size_t)M * DIM_;  // 16,777,216 elements
  const size_t WSZ = (size_t)DIM_ * DIM_;  // 1,048,576 elements

  char* ws = (char*)d_ws;
  bf16_t* Qb  = (bf16_t*)(ws);            // 2S
  bf16_t* Kb  = (bf16_t*)(ws + 2 * S);    // 2S
  bf16_t* Vb  = (bf16_t*)(ws + 4 * S);    // 2S
  bf16_t* qx  = (bf16_t*)(ws + 6 * S);    // 2S (reused as attended bf16 later)
  bf16_t* kx  = (bf16_t*)(ws + 8 * S);    // 2S
  bf16_t* vx  = (bf16_t*)(ws + 10 * S);   // 2S
  bf16_t* Wqb = (bf16_t*)(ws + 12 * S);
  bf16_t* Wkb = (bf16_t*)(ws + 12 * S + 2 * WSZ);
  bf16_t* Wvb = (bf16_t*)(ws + 12 * S + 4 * WSZ);
  bf16_t* Wob = (bf16_t*)(ws + 12 * S + 6 * WSZ);
  bf16_t* Ctx = (bf16_t*)(ws + 12 * S + 8 * WSZ);  // B*H*64*64*2 = 512 KB
  bf16_t* Ab  = qx;  // alias: qx dead after Q projection

  float* out = (float*)d_out;

  // fp32 -> bf16 conversions (activations + weights)
  const int gS = (int)(S / (256 * 8));    // 8192
  const int gW = (int)(WSZ / (256 * 8));  // 512
  to_bf16<<<gS, 256, 0, stream>>>(q_seq, qx);
  to_bf16<<<gS, 256, 0, stream>>>(k_seq, kx);
  to_bf16<<<gS, 256, 0, stream>>>(v_seq, vx);
  to_bf16<<<gW, 256, 0, stream>>>(Wq, Wqb);
  to_bf16<<<gW, 256, 0, stream>>>(Wk, Wkb);
  to_bf16<<<gW, 256, 0, stream>>>(Wv, Wvb);
  to_bf16<<<gW, 256, 0, stream>>>(Wo, Wob);

  dim3 gProj(DIM_ / 128, M / 128);  // (8, 128)
  gemm_bf16k<true><<<gProj, 256, 0, stream>>>(qx, Wqb, bq, (void*)Qb, M, DIM_, DIM_);
  gemm_bf16k<true><<<gProj, 256, 0, stream>>>(kx, Wkb, bk, (void*)Kb, M, DIM_, DIM_);
  gemm_bf16k<true><<<gProj, 256, 0, stream>>>(vx, Wvb, bv, (void*)Vb, M, DIM_, DIM_);

  softmax_col<<<dim3(DIM_, B_), 256, 0, stream>>>(Kb);

  context_kernel<<<dim3(NH_, B_), 128, 0, stream>>>(Kb, Vb, Ctx);

  attended_kernel<<<dim3(L_ / 64, B_ * NH_), 128, 0, stream>>>(Qb, Ctx, Ab);

  gemm_bf16k<false><<<gProj, 256, 0, stream>>>(Ab, Wob, bo, (void*)out, M, DIM_, DIM_);
}
